// MultiHeadAttention_1314259993135
// MI455X (gfx1250) — matrix operand order repaired and checked by reading, not compiled, not run on hardware
//
#include <hip/hip_runtime.h>
#include <hip/hip_bf16.h>

// ---------------------------------------------------------------------------
// MI455X (gfx1250) multi-head attention, bf16 WMMA path.
//   EMBED=1024, HEADS=16, HEAD_DIM=64, N=2, L=2048
// ---------------------------------------------------------------------------

#define EMBED    1024
#define HEADS    16
#define HEAD_DIM 64
#define NBATCH   2
#define SEQ      2048
#define MTOT     (NBATCH * SEQ)   // 4096 rows of activations

typedef __attribute__((ext_vector_type(16))) __bf16 v16bf;
typedef __attribute__((ext_vector_type(8)))  __bf16 bf16x8;
typedef __attribute__((ext_vector_type(8)))  float  v8f;

static __device__ __forceinline__ v8f wmma_bf16(v16bf a, v16bf b, v8f c) {
  // D = A(16x32 bf16) * B(32x16 bf16) + C(16x16 f32)
  return __builtin_amdgcn_wmma_f32_16x16x32_bf16(
      /*neg_a=*/false, a, /*neg_b=*/false, b,
      /*c_mod=*/(short)0, c, /*reuse_a=*/false, /*reuse_b=*/false);
}

// A-matrix fragment (16x32), row-major source, row stride in elements.
// ISA 16-bit A layout: lanes 0-15 hold K=[0..7]&[16..23], lanes 16-31 K=[8..15]&[24..31].
static __device__ __forceinline__ v16bf load_frag_a(const __bf16* __restrict__ base,
                                                    int row0, int row_stride, int kbase) {
  const int lane = threadIdx.x & 31;
  const __bf16* p = base + (size_t)(row0 + (lane & 15)) * row_stride
                         + kbase + ((lane & 16) ? 8 : 0);
  bf16x8 lo = *(const bf16x8*)p;        // global_load_b128 / ds_load_b128
  bf16x8 hi = *(const bf16x8*)(p + 16);
  return __builtin_shufflevector(lo, hi, 0,1,2,3,4,5,6,7,8,9,10,11,12,13,14,15);
}

// B-matrix fragment (32x16). Source holds B^T row-major (row n = column n of B).
// ISA 16-bit B layout: lanes 0-15 hold K=[0..7]&[16..23], lanes 16-31 K=[8..15]&[24..31].
static __device__ __forceinline__ v16bf load_frag_b(const __bf16* __restrict__ base,
                                                    int row0, int row_stride, int kbase) {
  const int lane = threadIdx.x & 31;
  const __bf16* p = base + (size_t)(row0 + (lane & 15)) * row_stride
                         + kbase + ((lane & 16) ? 8 : 0);
  bf16x8 lo = *(const bf16x8*)p;
  bf16x8 hi = *(const bf16x8*)(p + 16);
  return __builtin_shufflevector(lo, hi, 0,1,2,3,4,5,6,7,8,9,10,11,12,13,14,15);
}

// ---------------------------------------------------------------------------
// fp32 -> bf16 conversion
// ---------------------------------------------------------------------------
__global__ __launch_bounds__(256) void cvt_f32_bf16(const float* __restrict__ in,
                                                    __bf16* __restrict__ out, int n) {
  int i = blockIdx.x * 256 + threadIdx.x;
  if (i < n) out[i] = (__bf16)in[i];
}

// ---------------------------------------------------------------------------
// GEMM: C[m,e] = sum_k A16[m,k] * B16[e,k]  (i.e. A @ B^T), M=4096, K=N=1024.
//   mode 0: store bf16 into per-head layout  [b][h][l][d]
//   mode 1: store bf16 into transposed head  [b][h][d][l]   (for V)
//   mode 2: store fp32 [m][e] + bias[e]      (final output)
// Block: 256 threads (8 wave32), tile 128x64; wave tile 32x32 (2x2 WMMA).
// ---------------------------------------------------------------------------
__global__ __launch_bounds__(256) void gemm_bf16_kernel(const __bf16* __restrict__ A,
                                                        const __bf16* __restrict__ B,
                                                        void* __restrict__ Cout,
                                                        const float* __restrict__ bias,
                                                        int mode) {
  const int bm = blockIdx.x >> 4;      // 32 M-blocks
  const int bn = blockIdx.x & 15;      // 16 N-blocks
  const int wid = threadIdx.x >> 5;
  const int wm = wid & 3;
  const int wn = wid >> 2;
  const int row0 = bm * 128 + wm * 32;
  const int col0 = bn * 64 + wn * 32;
  const __bf16* Ab = A + (size_t)row0 * EMBED;
  const __bf16* Bb = B + (size_t)col0 * EMBED;

  v8f c00 = {}, c01 = {}, c10 = {}, c11 = {};
  for (int k = 0; k < EMBED; k += 32) {
    if (k + 32 < EMBED) {  // speculative prefetch of next K-slab (global_prefetch_b8)
      __builtin_prefetch(Ab + (size_t)(threadIdx.x & 15) * EMBED + k + 32, 0, 1);
      __builtin_prefetch(Bb + (size_t)(threadIdx.x & 15) * EMBED + k + 32, 0, 1);
    }
    v16bf a0 = load_frag_a(Ab, 0,  EMBED, k);
    v16bf a1 = load_frag_a(Ab, 16, EMBED, k);
    v16bf b0 = load_frag_b(Bb, 0,  EMBED, k);
    v16bf b1 = load_frag_b(Bb, 16, EMBED, k);
    c00 = wmma_bf16(a0, b0, c00);
    c01 = wmma_bf16(a0, b1, c01);
    c10 = wmma_bf16(a1, b0, c10);
    c11 = wmma_bf16(a1, b1, c11);
  }

  const int lane = threadIdx.x & 31;
  const int nsub = lane & 15;           // C/D layout: lane -> column
  const int mh   = (lane >> 4) & 1;     // lane half -> row offset 0/8
  v8f acc[2][2] = {{c00, c01}, {c10, c11}};
#pragma unroll
  for (int ti = 0; ti < 2; ++ti)
#pragma unroll
    for (int tj = 0; tj < 2; ++tj)
#pragma unroll
      for (int r = 0; r < 8; ++r) {
        int m = row0 + ti * 16 + mh * 8 + r;
        int e = col0 + tj * 16 + nsub;
        float v = acc[ti][tj][r];
        if (mode == 2) {
          ((float*)Cout)[(size_t)m * EMBED + e] = v + bias[e];
        } else {
          int b = m >> 11, l = m & (SEQ - 1);
          int h = e >> 6,  d = e & 63;
          size_t idx = (mode == 0)
              ? ((((size_t)b * HEADS + h) * SEQ + l) * HEAD_DIM + d)
              : ((((size_t)b * HEADS + h) * HEAD_DIM + d) * SEQ + l);
          ((__bf16*)Cout)[idx] = (__bf16)v;
        }
      }
}

// ---------------------------------------------------------------------------
// Flash attention. grid = (N*H) * (L/64) blocks, 128 threads (4 wave32).
// Each wave: 16 q-rows, online softmax over all 2048 keys in chunks of 32.
// Q[b,h,l,d], K[b,h,l,d], Vt[b,h,d,l] bf16; O[b,l,e] bf16.
// ---------------------------------------------------------------------------
__global__ __launch_bounds__(128) void attn_kernel(const __bf16* __restrict__ Q,
                                                   const __bf16* __restrict__ K,
                                                   const __bf16* __restrict__ Vt,
                                                   const int* __restrict__ mask,
                                                   __bf16* __restrict__ O) {
  __shared__ __align__(16) __bf16 ldsP[4][16][32];   // per-wave 16x32 P staging

  const int bh  = blockIdx.x >> 5;   // (b,h); SEQ/64 = 32 q-tiles
  const int qt  = blockIdx.x & 31;
  const int b   = bh >> 4;           // HEADS = 16
  const int h   = bh & 15;
  const int w   = threadIdx.x >> 5;
  const int lane = threadIdx.x & 31;
  const int nsub = lane & 15;
  const int mh   = (lane >> 4) & 1;
  const int qrow = qt * 64 + w * 16;

  const __bf16* Qb = Q  + ((size_t)bh * SEQ + qrow) * HEAD_DIM;
  const __bf16* Kb = K  + (size_t)bh * SEQ * HEAD_DIM;
  const __bf16* Vb = Vt + (size_t)bh * HEAD_DIM * SEQ;
  const int*    mb = mask + b * SEQ;

  const v16bf qf0 = load_frag_a(Qb, 0, HEAD_DIM, 0);    // d 0..31
  const v16bf qf1 = load_frag_a(Qb, 0, HEAD_DIM, 32);   // d 32..63

  v8f o0 = {}, o1 = {}, o2 = {}, o3 = {};
  float runmax[8], runsum[8];
#pragma unroll
  for (int r = 0; r < 8; ++r) { runmax[r] = -1e30f; runsum[r] = 0.0f; }
  const float scale = 0.125f;  // 1/sqrt(HEAD_DIM)
  v8f zero = {};

  for (int kc = 0; kc < SEQ; kc += 32) {
    // ---- S = (Q K^T) for this 32-key chunk: two 16x16 tiles, 4 WMMAs
    v16bf kf;
    kf = load_frag_b(Kb, kc,      HEAD_DIM, 0);  v8f s0 = wmma_bf16(qf0, kf, zero);
    kf = load_frag_b(Kb, kc,      HEAD_DIM, 32);     s0 = wmma_bf16(qf1, kf, s0);
    kf = load_frag_b(Kb, kc + 16, HEAD_DIM, 0);  v8f s1 = wmma_bf16(qf0, kf, zero);
    kf = load_frag_b(Kb, kc + 16, HEAD_DIM, 32);     s1 = wmma_bf16(qf1, kf, s1);

    const bool mz0 = (mb[kc + nsub] == 0);
    const bool mz1 = (mb[kc + 16 + nsub] == 0);

    // ---- online softmax: row max/sum via shfl_xor over 16-lane halves
    float al[8];
#pragma unroll
    for (int r = 0; r < 8; ++r) {
      float x0 = mz0 ? -1e30f : s0[r] * scale;
      float x1 = mz1 ? -1e30f : s1[r] * scale;
      float m0 = x0, m1 = x1;
#pragma unroll
      for (int dx = 1; dx < 16; dx <<= 1) {
        m0 = fmaxf(m0, __shfl_xor(m0, dx, 32));
        m1 = fmaxf(m1, __shfl_xor(m1, dx, 32));
      }
      float nm = fmaxf(runmax[r], fmaxf(m0, m1));
      float a  = __expf(runmax[r] - nm);
      float p0 = __expf(x0 - nm);
      float p1 = __expf(x1 - nm);
      s0[r] = p0; s1[r] = p1;
      float rs = p0 + p1;
#pragma unroll
      for (int dx = 1; dx < 16; dx <<= 1) rs += __shfl_xor(rs, dx, 32);
      runsum[r] = runsum[r] * a + rs;
      runmax[r] = nm;
      al[r] = a;
    }
#pragma unroll
    for (int r = 0; r < 8; ++r) {
      o0[r] *= al[r]; o1[r] *= al[r]; o2[r] *= al[r]; o3[r] *= al[r];
    }

    // ---- D-layout P -> bf16 A-fragment via wave-private LDS staging
    __syncthreads();
#pragma unroll
    for (int r = 0; r < 8; ++r) {
      ldsP[w][mh * 8 + r][nsub]      = (__bf16)s0[r];
      ldsP[w][mh * 8 + r][16 + nsub] = (__bf16)s1[r];
    }
    __syncthreads();
    v16bf pf = load_frag_a(&ldsP[w][0][0], 0, 32, 0);   // P 16x32 (q x keys)

    // ---- O += P @ V  (Vt rows are d, contiguous over keys): 4 WMMAs
    v16bf vf;
    vf = load_frag_b(Vb, 0,  SEQ, kc);  o0 = wmma_bf16(pf, vf, o0);
    vf = load_frag_b(Vb, 16, SEQ, kc);  o1 = wmma_bf16(pf, vf, o1);
    vf = load_frag_b(Vb, 32, SEQ, kc);  o2 = wmma_bf16(pf, vf, o2);
    vf = load_frag_b(Vb, 48, SEQ, kc);  o3 = wmma_bf16(pf, vf, o3);
  }

  // ---- normalize, store bf16 into [b][l][EMBED] for the output projection
#pragma unroll
  for (int r = 0; r < 8; ++r) {
    float inv = 1.0f / runsum[r];
    int q = qrow + mh * 8 + r;
    size_t off = ((size_t)b * SEQ + q) * EMBED + h * HEAD_DIM + nsub;
    O[off + 0]  = (__bf16)(o0[r] * inv);
    O[off + 16] = (__bf16)(o1[r] * inv);
    O[off + 32] = (__bf16)(o2[r] * inv);
    O[off + 48] = (__bf16)(o3[r] * inv);
  }
}

// ---------------------------------------------------------------------------
// Launch. Inputs: values, keys, query, mask(int), W_q, W_k, W_v, W_o, b_o.
// Workspace layout (bf16 elements): 64 MB total.
// ---------------------------------------------------------------------------
extern "C" void kernel_launch(void* const* d_in, const int* in_sizes, int n_in,
                              void* d_out, int out_size, void* d_ws, size_t ws_size,
                              hipStream_t stream) {
  const float* values = (const float*)d_in[0];
  const float* keys   = (const float*)d_in[1];
  const float* query  = (const float*)d_in[2];
  const int*   mask   = (const int*)d_in[3];
  const float* Wq = (const float*)d_in[4];
  const float* Wk = (const float*)d_in[5];
  const float* Wv = (const float*)d_in[6];
  const float* Wo = (const float*)d_in[7];
  const float* bo = (const float*)d_in[8];

  const size_t ACT = (size_t)MTOT * EMBED;   // 4096*1024
  const size_t WEL = (size_t)EMBED * EMBED;  // 1024*1024

  __bf16* Xq   = (__bf16*)d_ws;
  __bf16* Xk   = Xq   + ACT;
  __bf16* Xv   = Xk   + ACT;
  __bf16* Wq16 = Xv   + ACT;
  __bf16* Wk16 = Wq16 + WEL;
  __bf16* Wv16 = Wk16 + WEL;
  __bf16* Wo16 = Wv16 + WEL;
  __bf16* Q16  = Wo16 + WEL;   // [b][h][l][d]
  __bf16* K16  = Q16  + ACT;   // [b][h][l][d]
  __bf16* Vt16 = K16  + ACT;   // [b][h][d][l]
  __bf16* O16  = Vt16 + ACT;   // [b][l][e]

  const int na = (int)ACT, nw = (int)WEL;
  cvt_f32_bf16<<<(na + 255) / 256, 256, 0, stream>>>(query,  Xq, na);
  cvt_f32_bf16<<<(na + 255) / 256, 256, 0, stream>>>(keys,   Xk, na);
  cvt_f32_bf16<<<(na + 255) / 256, 256, 0, stream>>>(values, Xv, na);
  cvt_f32_bf16<<<(nw + 255) / 256, 256, 0, stream>>>(Wq, Wq16, nw);
  cvt_f32_bf16<<<(nw + 255) / 256, 256, 0, stream>>>(Wk, Wk16, nw);
  cvt_f32_bf16<<<(nw + 255) / 256, 256, 0, stream>>>(Wv, Wv16, nw);
  cvt_f32_bf16<<<(nw + 255) / 256, 256, 0, stream>>>(Wo, Wo16, nw);

  // Projections: Q,K per-head; V per-head transposed.
  gemm_bf16_kernel<<<512, 256, 0, stream>>>(Xq, Wq16, Q16,  nullptr, 0);
  gemm_bf16_kernel<<<512, 256, 0, stream>>>(Xk, Wk16, K16,  nullptr, 0);
  gemm_bf16_kernel<<<512, 256, 0, stream>>>(Xv, Wv16, Vt16, nullptr, 1);

  // Flash attention: (N*H) * (L/64) = 1024 blocks of 128 threads.
  attn_kernel<<<(NBATCH * HEADS) * (SEQ / 64), 128, 0, stream>>>(Q16, K16, Vt16, mask, O16);

  // Output projection with bias -> fp32.
  gemm_bf16_kernel<<<512, 256, 0, stream>>>(O16, Wo16, d_out, bo, 2);
}
